// RoPEBidirectionalAttention_5188320493801
// MI455X (gfx1250) — compile-verified, matrix operand order checked by reading, not run on hardware
//
#include <hip/hip_runtime.h>
#include <hip/hip_bf16.h>

typedef __bf16 bf16;
typedef bf16  v16bf __attribute__((ext_vector_type(16)));
typedef float v8f   __attribute__((ext_vector_type(8)));
typedef unsigned int u32x4 __attribute__((ext_vector_type(4)));
typedef int          i32x4 __attribute__((ext_vector_type(4)));
typedef int          i32x8 __attribute__((ext_vector_type(8)));

#if __has_builtin(__builtin_amdgcn_tensor_load_to_lds) && __has_builtin(__builtin_amdgcn_s_wait_tensorcnt)
#define HAVE_TDM 1
#else
#define HAVE_TDM 0
#endif

union Frag {
    v16bf v;
    uint4 u[2];
};

static __device__ __forceinline__ v8f wmma_bf16(const Frag& a, const Frag& b, v8f c) {
    return __builtin_amdgcn_wmma_f32_16x16x32_bf16(
        /*neg_a=*/false, a.v, /*neg_b=*/false, b.v,
        /*c_mod=*/(short)0, c, /*reuse_a=*/false, /*reuse_b=*/false);
}

#if HAVE_TDM
// Issue a TDM 2D tile load: tile_y rows x tile_x bf16 elements, row stride
// row_stride_elems, from gaddr into LDS byte offset lds_off (packed rows).
// D# layout per CDNA5 ISA (group0 128b, group1 256b).
static __device__ __forceinline__ void tdm_load_2d(
    const void* gaddr, unsigned lds_off,
    unsigned tile_x, unsigned tile_y,
    unsigned tensor_x, unsigned tensor_y,
    unsigned long long row_stride_elems)
{
    unsigned long long ga = (unsigned long long)(uintptr_t)gaddr;
    u32x4 g0;
    g0[0] = 1u;                                                // count=1, user D#
    g0[1] = lds_off;                                           // lds_addr
    g0[2] = (unsigned)(ga & 0xFFFFFFFFu);                      // global_addr[31:0]
    g0[3] = (unsigned)((ga >> 32) & 0x01FFFFFFu) | (2u << 30); // ga[56:32] | type=2
    i32x8 g1;
    g1[0] = (int)(1u << 16);                                   // data_size=1 (2 bytes)
    g1[1] = (int)((tensor_x & 0xFFFFu) << 16);                 // tensor_dim0[15:0]
    g1[2] = (int)(((tensor_x >> 16) & 0xFFFFu) |
                  ((tensor_y & 0xFFFFu) << 16));               // td0[31:16] | td1[15:0]
    g1[3] = (int)(((tensor_y >> 16) & 0xFFFFu) |
                  ((tile_x & 0xFFFFu) << 16));                 // td1[31:16] | tile_dim0
    g1[4] = (int)(tile_y & 0xFFFFu);                           // tile_dim1, tile_dim2=0
    g1[5] = (int)(unsigned)(row_stride_elems & 0xFFFFFFFFull); // td0_stride[31:0]
    g1[6] = (int)(unsigned)((row_stride_elems >> 32) & 0xFFFFull);
    g1[7] = 0;
    i32x4 z4 = {0, 0, 0, 0};
#if defined(__clang_major__) && (__clang_major__ >= 23)
    i32x8 z8 = {0, 0, 0, 0, 0, 0, 0, 0};
    __builtin_amdgcn_tensor_load_to_lds(g0, g1, z4, z4, z8, 0);
#else
    __builtin_amdgcn_tensor_load_to_lds(g0, g1, z4, z4, 0);
#endif
}
#endif

// ---------------------------------------------------------------------------
// Elementwise f32 -> bf16 convert
// ---------------------------------------------------------------------------
__global__ void cvt_f32_bf16(const float* __restrict__ in, bf16* __restrict__ out, size_t n) {
    size_t i = (size_t)blockIdx.x * blockDim.x + threadIdx.x;
    if (i < n) out[i] = (bf16)in[i];
}

// Transpose (K,N) f32 -> (N,K) bf16 so GEMM B-fragments are K-contiguous.
__global__ void cvt_transpose_f32_bf16(const float* __restrict__ in, bf16* __restrict__ out,
                                       int K, int N) {
    size_t i = (size_t)blockIdx.x * blockDim.x + threadIdx.x;
    size_t total = (size_t)K * N;
    if (i >= total) return;
    int k = (int)(i % K);
    int n = (int)(i / K);
    out[(size_t)n * K + k] = (bf16)in[(size_t)k * N + n];
}

// ---------------------------------------------------------------------------
// bf16 GEMM: C(M,N) = A(M,K) * Bt(N,K)^T, f32 accumulate via WMMA.
// Block tile 128x128, K-step 32, 8 waves, wave tile 64x32 (4x2 WMMA tiles).
// LDS staging is double-buffered through the Tensor Data Mover: wave 0 issues
// tensor_load_to_lds for tile i+1 while all waves run WMMAs on tile i, then
// s_wait_tensorcnt(2) (TDM completes in order) before the barrier.
// ---------------------------------------------------------------------------
template <bool OUT_F32>
__global__ __launch_bounds__(256) void gemm_bf16_wmma(
    const bf16* __restrict__ A, const bf16* __restrict__ Bt,
    void* __restrict__ Cout, int M, int N, int K) {

    __shared__ __align__(16) bf16 sA[2][128 * 32];
    __shared__ __align__(16) bf16 sB[2][128 * 32];

    const int tid  = threadIdx.x;
    const int wave = tid >> 5;
    const int lane = tid & 31;
    const int half = lane >> 4;
    const int l15  = lane & 15;

    const int brow = blockIdx.y * 128;
    const int bcol = blockIdx.x * 128;
    const int wrow = (wave >> 2) * 64;   // 2 wave rows of 64
    const int wcol = (wave & 3) * 32;    // 4 wave cols of 32

    v8f acc[4][2];
#pragma unroll
    for (int mt = 0; mt < 4; ++mt)
#pragma unroll
        for (int nt = 0; nt < 2; ++nt) acc[mt][nt] = {};

#if HAVE_TDM
    // Prologue: wave 0 kicks off TDM loads of the first K-tile into buffer 0.
    if (wave == 0) {
        tdm_load_2d(&A [(size_t)brow * K], (unsigned)(uintptr_t)&sA[0][0],
                    32, 128, (unsigned)K, (unsigned)M, (unsigned long long)K);
        tdm_load_2d(&Bt[(size_t)bcol * K], (unsigned)(uintptr_t)&sB[0][0],
                    32, 128, (unsigned)K, (unsigned)N, (unsigned long long)K);
    }
    int buf = 0;
    for (int k0 = 0; k0 < K; k0 += 32) {
        if (wave == 0) {
            if (k0 + 32 < K) {
                // Prefetch next tile into the other buffer, keep it in flight.
                tdm_load_2d(&A [(size_t)brow * K + k0 + 32],
                            (unsigned)(uintptr_t)&sA[buf ^ 1][0],
                            32, 128, (unsigned)K, (unsigned)M, (unsigned long long)K);
                tdm_load_2d(&Bt[(size_t)bcol * K + k0 + 32],
                            (unsigned)(uintptr_t)&sB[buf ^ 1][0],
                            32, 128, (unsigned)K, (unsigned)N, (unsigned long long)K);
                __builtin_amdgcn_s_wait_tensorcnt(2);  // current tile resident
            } else {
                __builtin_amdgcn_s_wait_tensorcnt(0);  // drain for last tile
            }
        }
        __syncthreads();
        const bf16* tA = &sA[buf][0];
        const bf16* tB = &sB[buf][0];
#else
    for (int k0 = 0; k0 < K; k0 += 32) {
        // Fallback staging: per-thread b128 global->LDS copies.
#pragma unroll
        for (int it = 0; it < 2; ++it) {
            int i = tid + it * 256;          // 512 uint4 per tile
            int r = i >> 2;                  // 4 uint4 per 32-bf16 row
            int c = (i & 3) * 8;
            *(uint4*)&sA[0][r * 32 + c] = *(const uint4*)&A [(size_t)(brow + r) * K + k0 + c];
            *(uint4*)&sB[0][r * 32 + c] = *(const uint4*)&Bt[(size_t)(bcol + r) * K + k0 + c];
        }
        __syncthreads();
        if (k0 + 32 < K) {
            __builtin_prefetch(&A [(size_t)(brow + (tid >> 1)) * K + k0 + 32], 0, 0);
            __builtin_prefetch(&Bt[(size_t)(bcol + (tid >> 1)) * K + k0 + 32], 0, 0);
        }
        const bf16* tA = &sA[0][0];
        const bf16* tB = &sB[0][0];
#endif

        // Fragments: lanes 0-15 hold K 0-7 & 16-23, lanes 16-31 hold K 8-15 & 24-31.
        Frag a[4], b[2];
#pragma unroll
        for (int mt = 0; mt < 4; ++mt) {
            int m = wrow + mt * 16 + l15;
            a[mt].u[0] = *(const uint4*)&tA[m * 32 + half * 8];
            a[mt].u[1] = *(const uint4*)&tA[m * 32 + 16 + half * 8];
        }
#pragma unroll
        for (int nt = 0; nt < 2; ++nt) {
            int n = wcol + nt * 16 + l15;
            b[nt].u[0] = *(const uint4*)&tB[n * 32 + half * 8];
            b[nt].u[1] = *(const uint4*)&tB[n * 32 + 16 + half * 8];
        }
#pragma unroll
        for (int mt = 0; mt < 4; ++mt)
#pragma unroll
            for (int nt = 0; nt < 2; ++nt)
                acc[mt][nt] = wmma_bf16(a[mt], b[nt], acc[mt][nt]);
        __syncthreads();
#if HAVE_TDM
        buf ^= 1;
#endif
    }

    // Epilogue. C layout: VGPR r -> row (r + half*8), column = lane&15.
#pragma unroll
    for (int mt = 0; mt < 4; ++mt) {
#pragma unroll
        for (int nt = 0; nt < 2; ++nt) {
#pragma unroll
            for (int r = 0; r < 8; ++r) {
                int m = brow + wrow + mt * 16 + half * 8 + r;
                int n = bcol + wcol + nt * 16 + l15;
                float v = acc[mt][nt][r];
                if (OUT_F32) ((float*)Cout)[(size_t)m * N + n] = v;
                else         ((bf16*) Cout)[(size_t)m * N + n] = (bf16)v;
            }
        }
    }
}

// ---------------------------------------------------------------------------
// RoPE + split qkv (B*T, 3C) bf16 -> q,k,v as (B,H,T,D) bf16. One thread per
// rotate pair (d, d+32).
// ---------------------------------------------------------------------------
__global__ void rope_split(const bf16* __restrict__ qkv,
                           bf16* __restrict__ q, bf16* __restrict__ k, bf16* __restrict__ v,
                           int B, int T, int H, int Dh) {
    const int Dh2 = Dh / 2;
    size_t idx = (size_t)blockIdx.x * blockDim.x + threadIdx.x;
    size_t total = (size_t)B * H * T * Dh2;
    if (idx >= total) return;

    int d = (int)(idx % Dh2); size_t rest = idx / Dh2;
    int t = (int)(rest % T);  rest /= T;
    int h = (int)(rest % H);
    int b = (int)(rest / H);

    const int C3 = 3 * H * Dh;
    size_t inBase = ((size_t)b * T + t) * C3 + h * Dh;
    size_t outBase = (((size_t)b * H + h) * (size_t)T + t) * Dh;

    float ang = (float)t * __powf(10000.0f, -(2.0f * d) / (float)Dh);
    float cs = __cosf(ang), sn = __sinf(ang);

    float q1 = (float)qkv[inBase + d], q2 = (float)qkv[inBase + d + Dh2];
    q[outBase + d]       = (bf16)(q1 * cs - q2 * sn);
    q[outBase + d + Dh2] = (bf16)(q2 * cs + q1 * sn);

    float k1 = (float)qkv[inBase + H * Dh + d], k2 = (float)qkv[inBase + H * Dh + d + Dh2];
    k[outBase + d]       = (bf16)(k1 * cs - k2 * sn);
    k[outBase + d + Dh2] = (bf16)(k2 * cs + k1 * sn);

    v[outBase + d]       = qkv[inBase + 2 * H * Dh + d];
    v[outBase + d + Dh2] = qkv[inBase + 2 * H * Dh + d + Dh2];
}

// ---------------------------------------------------------------------------
// Flash attention, bidirectional, D=64. One block = 128 query rows of one
// (b,h); 8 waves, each wave owns 16 query rows. Key chunks of 64 through LDS.
// Output written directly in (B,T,H*D) layout for the projection GEMM.
// ---------------------------------------------------------------------------
__global__ __launch_bounds__(256) void flash_attn_wmma(
    const bf16* __restrict__ qg, const bf16* __restrict__ kg, const bf16* __restrict__ vg,
    bf16* __restrict__ yb, int B, int T, int H) {

    constexpr int Dh = 64;
    __shared__ __align__(16) bf16 sK[64 * Dh];       // (key, d)    8 KB
    __shared__ __align__(16) bf16 sV[Dh * 64];       // (d, key)    8 KB
    __shared__ __align__(16) bf16 sP[8][16 * 64];    // per-wave P 16 KB

    const int tid  = threadIdx.x;
    const int wave = tid >> 5;
    const int lane = tid & 31;
    const int half = lane >> 4;
    const int l15  = lane & 15;

    const int bh = blockIdx.y;
    const int b  = bh / H;
    const int h  = bh % H;
    const int qbase = blockIdx.x * 128;

    const bf16* qptr = qg + ((size_t)bh) * T * Dh;
    const bf16* kptr = kg + ((size_t)bh) * T * Dh;
    const bf16* vptr = vg + ((size_t)bh) * T * Dh;

    // Preload this wave's Q tile (16 x 64) as two A fragments (K=0..31, 32..63).
    Frag qa[2];
    {
        int qm = qbase + wave * 16 + l15;
#pragma unroll
        for (int kk = 0; kk < 2; ++kk) {
            qa[kk].u[0] = *(const uint4*)&qptr[(size_t)qm * Dh + kk * 32 + half * 8];
            qa[kk].u[1] = *(const uint4*)&qptr[(size_t)qm * Dh + kk * 32 + 16 + half * 8];
        }
    }

    const float scale = 0.125f;   // 1/sqrt(64)
    float mrow[8], lrow[8];
    v8f o[4];
#pragma unroll
    for (int r = 0; r < 8; ++r) { mrow[r] = -1e30f; lrow[r] = 0.0f; }
#pragma unroll
    for (int nt = 0; nt < 4; ++nt) o[nt] = {};

    for (int kc = 0; kc < T; kc += 64) {
        // Stage K chunk (64x64) row-major and V chunk transposed to (d,key).
#pragma unroll
        for (int it = 0; it < 2; ++it) {
            int i = tid + it * 256;           // 512 uint4
            int r = i >> 3;                   // 8 uint4 per 64-bf16 row
            int c = (i & 7) * 8;
            *(uint4*)&sK[r * Dh + c] = *(const uint4*)&kptr[(size_t)(kc + r) * Dh + c];
            union { uint4 u; bf16 e[8]; } tv;
            tv.u = *(const uint4*)&vptr[(size_t)(kc + r) * Dh + c];
#pragma unroll
            for (int j = 0; j < 8; ++j) sV[(c + j) * 64 + r] = tv.e[j];
        }
        __syncthreads();
        if (kc + 64 < T) {
            __builtin_prefetch(&kptr[(size_t)(kc + 64 + (tid >> 2)) * Dh], 0, 0);
            __builtin_prefetch(&vptr[(size_t)(kc + 64 + (tid >> 2)) * Dh], 0, 0);
        }

        // Scores: 4 key tiles of 16, each = 2 WMMAs over D.
        v8f S[4];
#pragma unroll
        for (int jt = 0; jt < 4; ++jt) {
            Frag bk0, bk1;
            int kn = jt * 16 + l15;
            bk0.u[0] = *(const uint4*)&sK[kn * Dh + half * 8];
            bk0.u[1] = *(const uint4*)&sK[kn * Dh + 16 + half * 8];
            bk1.u[0] = *(const uint4*)&sK[kn * Dh + 32 + half * 8];
            bk1.u[1] = *(const uint4*)&sK[kn * Dh + 48 + half * 8];
            v8f s = {};
            s = wmma_bf16(qa[0], bk0, s);
            s = wmma_bf16(qa[1], bk1, s);
            S[jt] = s;
        }

        // Online softmax. Row r lives across 16 lanes of one half.
#pragma unroll
        for (int r = 0; r < 8; ++r) {
            float mloc = -1e30f;
#pragma unroll
            for (int jt = 0; jt < 4; ++jt) mloc = fmaxf(mloc, S[jt][r] * scale);
#pragma unroll
            for (int off = 8; off > 0; off >>= 1)
                mloc = fmaxf(mloc, __shfl_xor(mloc, off, 16));
            float mnew  = fmaxf(mrow[r], mloc);
            float alpha = __expf(mrow[r] - mnew);
            float psum = 0.0f;
#pragma unroll
            for (int jt = 0; jt < 4; ++jt) {
                float p = __expf(S[jt][r] * scale - mnew);
                S[jt][r] = p;
                psum += p;
            }
#pragma unroll
            for (int off = 8; off > 0; off >>= 1)
                psum += __shfl_xor(psum, off, 16);
            lrow[r] = lrow[r] * alpha + psum;
            mrow[r] = mnew;
#pragma unroll
            for (int nt = 0; nt < 4; ++nt) o[nt][r] *= alpha;
        }

        // Re-stripe P from C-layout to A-layout via per-wave LDS scratch.
        bf16* pw = &sP[wave][0];
#pragma unroll
        for (int jt = 0; jt < 4; ++jt)
#pragma unroll
            for (int r = 0; r < 8; ++r)
                pw[(half * 8 + r) * 64 + jt * 16 + l15] = (bf16)S[jt][r];

        // O += P(16x64) * V(64x64): 2 K-steps x 4 N-tiles of WMMA.
#pragma unroll
        for (int kk = 0; kk < 2; ++kk) {
            Frag pa;
            pa.u[0] = *(const uint4*)&pw[l15 * 64 + kk * 32 + half * 8];
            pa.u[1] = *(const uint4*)&pw[l15 * 64 + kk * 32 + 16 + half * 8];
#pragma unroll
            for (int nt = 0; nt < 4; ++nt) {
                Frag bv;
                int dn = nt * 16 + l15;
                bv.u[0] = *(const uint4*)&sV[dn * 64 + kk * 32 + half * 8];
                bv.u[1] = *(const uint4*)&sV[dn * 64 + kk * 32 + 16 + half * 8];
                o[nt] = wmma_bf16(pa, bv, o[nt]);
            }
        }
        __syncthreads();
    }

    // Normalize and write O to (B, T, H*D) bf16 for the projection GEMM.
#pragma unroll
    for (int r = 0; r < 8; ++r) {
        float inv_l = 1.0f / lrow[r];
        int t = qbase + wave * 16 + half * 8 + r;
#pragma unroll
        for (int nt = 0; nt < 4; ++nt) {
            int d = nt * 16 + l15;
            yb[((size_t)b * T + t) * (size_t)(H * Dh) + h * Dh + d] = (bf16)(o[nt][r] * inv_l);
        }
    }
}

// ---------------------------------------------------------------------------
// Host launcher
// ---------------------------------------------------------------------------
extern "C" void kernel_launch(void* const* d_in, const int* in_sizes, int n_in,
                              void* d_out, int out_size, void* d_ws, size_t ws_size,
                              hipStream_t stream) {
    constexpr int B = 4, T = 2048, C = 1024, H = 16, Dh = 64;
    constexpr int BT = B * T;          // 8192
    constexpr int C3 = 3 * C;          // 3072

    const float* x      = (const float*)d_in[0];
    const float* W_attn = (const float*)d_in[1];
    const float* W_proj = (const float*)d_in[2];
    float* out = (float*)d_out;

    char* ws = (char*)d_ws;
    size_t off = 0;
    bf16* xb   = (bf16*)(ws + off); off += (size_t)BT * C  * sizeof(bf16);   // 16 MB
    bf16* WaT  = (bf16*)(ws + off); off += (size_t)C3 * C  * sizeof(bf16);   //  6 MB
    bf16* WpT  = (bf16*)(ws + off); off += (size_t)C  * C  * sizeof(bf16);   //  2 MB
    bf16* qkv  = (bf16*)(ws + off); off += (size_t)BT * C3 * sizeof(bf16);   // 48 MB
    bf16* qB   = (bf16*)(ws + off); off += (size_t)BT * C  * sizeof(bf16);   // 16 MB
    bf16* kB   = (bf16*)(ws + off); off += (size_t)BT * C  * sizeof(bf16);   // 16 MB
    bf16* vB   = (bf16*)(ws + off); off += (size_t)BT * C  * sizeof(bf16);   // 16 MB
    bf16* yB   = (bf16*)(ws + off); off += (size_t)BT * C  * sizeof(bf16);   // 16 MB
    (void)ws_size; (void)n_in; (void)in_sizes; (void)out_size;

    // 1. Precision conversion (weights pre-transposed to (n,k)).
    {
        size_t n = (size_t)BT * C;
        cvt_f32_bf16<<<(unsigned)((n + 255) / 256), 256, 0, stream>>>(x, xb, n);
        size_t na = (size_t)C * C3;
        cvt_transpose_f32_bf16<<<(unsigned)((na + 255) / 256), 256, 0, stream>>>(W_attn, WaT, C, C3);
        size_t np = (size_t)C * C;
        cvt_transpose_f32_bf16<<<(unsigned)((np + 255) / 256), 256, 0, stream>>>(W_proj, WpT, C, C);
    }

    // 2. QKV GEMM: (8192 x 1024) * (1024 x 3072) -> bf16 qkv.
    gemm_bf16_wmma<false><<<dim3(C3 / 128, BT / 128), 256, 0, stream>>>(
        xb, WaT, qkv, BT, C3, C);

    // 3. RoPE + head split.
    {
        size_t n = (size_t)B * H * T * (Dh / 2);
        rope_split<<<(unsigned)((n + 255) / 256), 256, 0, stream>>>(qkv, qB, kB, vB, B, T, H, Dh);
    }

    // 4. Flash attention (bidirectional), writes (B,T,C) bf16.
    flash_attn_wmma<<<dim3(T / 128, B * H), 256, 0, stream>>>(qB, kB, vB, yB, B, T, H);

    // 5. Output projection: (8192 x 1024) * (1024 x 1024) -> f32 out.
    gemm_bf16_wmma<true><<<dim3(C / 128, BT / 128), 256, 0, stream>>>(
        yB, WpT, out, BT, C, C);
}
